// WindowAttention_66529043415701
// MI455X (gfx1250) — matrix operand order repaired and checked by reading, not compiled, not run on hardware
//
#include <hip/hip_runtime.h>

typedef __attribute__((ext_vector_type(16))) _Float16 v16h;
typedef __attribute__((ext_vector_type(4)))  _Float16 v4h;
typedef __attribute__((ext_vector_type(8)))  float    v8f;
typedef __attribute__((ext_vector_type(4)))  float    v4f;

#define NTOK  49
#define CDIM  128
#define NHEAD 4
#define HD    32
#define NPAD  64
#define NW    64

// ---- WMMA fragment helpers (CDNA5 wave32, 16x16x32 f16 -> f32) -------------
// A-frag (16x32, MxK): lane&15 = row M; lanes 0-15 hold K = k0+0..7 and
// k0+16..23; lanes 16-31 hold K = k0+8..15 and k0+24..31 (ISA 7.12.2).
__device__ __forceinline__ v16h load_a_frag(const _Float16* buf, int row0, int k0, int ld) {
  const int lane = threadIdx.x & 31;
  const _Float16* p = buf + (row0 + (lane & 15)) * ld + k0 + ((lane >> 4) << 3);
  v16h r;
  reinterpret_cast<v4f*>(&r)[0] = *reinterpret_cast<const v4f*>(p);
  reinterpret_cast<v4f*>(&r)[1] = *reinterpret_cast<const v4f*>(p + 16);
  return r;
}

// B-frag (32x16, KxN), source stored transposed as [N][K]: lane&15 = column N;
// lanes 0-15 hold K = k0+0..7 and k0+16..23; lanes 16-31 hold K = k0+8..15
// and k0+24..31 (same K order as the A-frag).
__device__ __forceinline__ v16h load_b_frag(const _Float16* buf, int col0, int k0, int ld) {
  const int lane = threadIdx.x & 31;
  const _Float16* p = buf + (col0 + (lane & 15)) * ld + k0 + ((lane >> 4) << 3);
  v16h r;
  reinterpret_cast<v4f*>(&r)[0] = *reinterpret_cast<const v4f*>(p);
  reinterpret_cast<v4f*>(&r)[1] = *reinterpret_cast<const v4f*>(p + 16);
  return r;
}

__device__ __forceinline__ v8f wmma16(v16h a, v16h b, v8f c) {
  // (neg_a, A, neg_b, B, c_mod, C, reuse_a, reuse_b)
  return __builtin_amdgcn_wmma_f32_16x16x32_f16(false, a, false, b, (short)0, c, false, false);
}

// ---- Prep: f16-transposed weights + fused (bias + mask) table, transposed --
// bmT[w][h][j][i] (64x4x64x64 f32): -1e30 for j>=49 (softmax mask), 0 for pad
// i, else rel-pos bias(h,i,j) + shift mask(w,i,j). Shared by all 4096 windows;
// transposed so the S^T-fragment add is one coalesced load per element.
__global__ void prep_kernel(const float* __restrict__ qkv_w, const float* __restrict__ proj_w,
                            const float* __restrict__ bias_table, const float* __restrict__ mask,
                            _Float16* __restrict__ wqkv_t, _Float16* __restrict__ wproj_t,
                            float* __restrict__ bmT) {
  int e = blockIdx.x * blockDim.x + threadIdx.x;
  if (e < 128 * 384) wqkv_t[(e % 384) * 128 + (e / 384)] = (_Float16)qkv_w[e];
  if (e < 128 * 128) wproj_t[(e % 128) * 128 + (e / 128)] = (_Float16)proj_w[e];
  {
    int w = e >> 14;            // /(4*64*64)
    int h = (e >> 12) & 3;
    int j = (e >> 6) & 63;      // middle index = key position
    int i = e & 63;             // minor index  = query position
    float v;
    if (j >= NTOK)      v = -1e30f;
    else if (i >= NTOK) v = 0.0f;
    else {
      int idx = 13 * ((i % 7) - (j % 7) + 6) + ((i / 7) - (j / 7) + 6);
      v = bias_table[idx * NHEAD + h] + mask[w * NTOK * NTOK + i * NTOK + j];
    }
    bmT[e] = v;
  }
}

// ---- Fused window attention: one workgroup (8 wave32) per window -----------
__global__ void __launch_bounds__(256)
win_attn_kernel(const float* __restrict__ x,
                const float* __restrict__ qkv_b, const float* __restrict__ proj_b,
                const float* __restrict__ bmT,
                const _Float16* __restrict__ wqkv_t, const _Float16* __restrict__ wproj_t,
                float* __restrict__ out) {
  extern __shared__ _Float16 smem[];
  _Float16* AO = smem;              // [64][128]: x(f16) in phase1, O(f16) in phase2/3
  _Float16* Qb = smem + 8192;       // [4][64][32], pre-scaled
  _Float16* Kb = smem + 16384;      // [4][64][32]
  _Float16* Vt = smem + 24576;      // [4][32][64] (transposed V)

  const int l = blockIdx.x;
  const int tid = threadIdx.x;
  const int wave = tid >> 5;
  const int lane = tid & 31;
  const int lane15 = lane & 15;
  const int hi8 = (lane >> 4) << 3;
  const float scale = 0.17677669529663687f; // hd^-0.5

  // Phase 0: stage x[l] as f16 (b128 NT loads -> b64 LDS stores), then pad
  const float* xl = x + (size_t)l * NTOK * CDIM;
  for (int e = tid; e < NTOK * (CDIM / 4); e += 256) {   // 1568 vec4 groups
    v4f f = __builtin_nontemporal_load(reinterpret_cast<const v4f*>(xl) + e);
    v4h hv;
    hv[0] = (_Float16)f[0]; hv[1] = (_Float16)f[1];
    hv[2] = (_Float16)f[2]; hv[3] = (_Float16)f[3];
    reinterpret_cast<v4h*>(AO)[e] = hv;
  }
  for (int e = tid; e < (NPAD - NTOK) * (CDIM / 4); e += 256) {  // 480 groups
    v4h z = {};
    reinterpret_cast<v4h*>(AO + NTOK * CDIM)[e] = z;
  }
  __syncthreads();

  // Phase 1: QKV GEMM [64,128] x [128,384]; 24 col-tiles / 8 waves.
  // B-frags hoisted per column tile (loaded once, reused by all 4 row tiles).
  for (int c = 0; c < 3; ++c) {
    const int ct = wave * 3 + c;
    v16h b0 = load_b_frag(wqkv_t, ct * 16, 0,  CDIM);
    v16h b1 = load_b_frag(wqkv_t, ct * 16, 32, CDIM);
    v16h b2 = load_b_frag(wqkv_t, ct * 16, 64, CDIM);
    v16h b3 = load_b_frag(wqkv_t, ct * 16, 96, CDIM);
    const int ctu = __builtin_amdgcn_readfirstlane(ct);   // wave-uniform tile id
    const int h   = (ctu >> 1) & 3;                       // head, uniform
    const int d   = ((ctu & 1) << 4) + lane15;            // 0..31 within head
    const float bv = qkv_b[ctu * 16 + lane15];
    for (int rt = 0; rt < 4; ++rt) {
      v8f acc = {};
      acc = wmma16(load_a_frag(AO, rt * 16, 0,  CDIM), b0, acc);
      acc = wmma16(load_a_frag(AO, rt * 16, 32, CDIM), b1, acc);
      acc = wmma16(load_a_frag(AO, rt * 16, 64, CDIM), b2, acc);
      acc = wmma16(load_a_frag(AO, rt * 16, 96, CDIM), b3, acc);
      const int rowb = rt * 16 + hi8;
      if (ctu < 8) {                                      // Q (pre-scaled)
        _Float16* q = Qb + h * 2048 + rowb * 32 + d;
        #pragma unroll
        for (int r = 0; r < 8; ++r) q[r * 32] = (_Float16)((acc[r] + bv) * scale);
      } else if (ctu < 16) {                              // K
        _Float16* k = Kb + h * 2048 + rowb * 32 + d;
        #pragma unroll
        for (int r = 0; r < 8; ++r) k[r * 32] = (_Float16)(acc[r] + bv);
      } else {                                            // V, transposed
        _Float16* vp = Vt + h * 2048 + d * 64 + rowb;
        #pragma unroll
        for (int r = 0; r < 8; ++r) vp[r] = (_Float16)(acc[r] + bv);
      }
    }
  }
  __syncthreads();

  // Phase 2: attention; 16 (head,row-tile) combos, 2 per wave.
  // Compute S^T = K x Q^T so each softmax row i lives in lane pair (L, L^16):
  // per-lane reductions + ONE shfl, and the normalized P is already in WMMA
  // A-fragment layout for O = P x V (no LDS round-trip).
  for (int combo = wave * 2; combo < wave * 2 + 2; ++combo) {
    const int h = combo >> 2, it = combo & 3;
    const _Float16* Qh = Qb + h * 2048;
    const _Float16* Kh = Kb + h * 2048;
    const _Float16* Vh = Vt + h * 2048;

    v16h bq = load_b_frag(Qh, it * 16, 0, HD);   // Q^T as B (Qb is row-major)
    v8f s[4];
    #pragma unroll
    for (int jt = 0; jt < 4; ++jt) {
      v8f z = {};
      s[jt] = wmma16(load_a_frag(Kh, jt * 16, 0, HD), bq, z);  // S^T tile
    }

    // fused rel-pos bias + shift mask + padding mask (transposed, coalesced:
    // one base address + 32 immediate offsets)
    const float* bmc = bmT + ((size_t)(l & (NW - 1)) * NHEAD + h) * (NPAD * NPAD)
                     + it * 16 + lane15;
    #pragma unroll
    for (int jt = 0; jt < 4; ++jt)
      #pragma unroll
      for (int r = 0; r < 8; ++r)
        s[jt][r] += bmc[(jt * 16 + r + hi8) << 6];

    // per-lane softmax over j: this lane holds 32 of row i's 64 entries,
    // lane^16 holds the rest
    float m = s[0][0];
    #pragma unroll
    for (int jt = 0; jt < 4; ++jt)
      #pragma unroll
      for (int r = 0; r < 8; ++r) m = fmaxf(m, s[jt][r]);
    m = fmaxf(m, __shfl_xor(m, 16, 32));
    float sum = 0.0f;
    #pragma unroll
    for (int jt = 0; jt < 4; ++jt)
      #pragma unroll
      for (int r = 0; r < 8; ++r) { float e = __expf(s[jt][r] - m); s[jt][r] = e; sum += e; }
    sum += __shfl_xor(sum, 16, 32);
    const float inv = 1.0f / sum;

    // P in registers IS the A-fragment: frag kt element e = s[2kt][e], e+8 = s[2kt+1][e]
    v16h p0, p1;
    #pragma unroll
    for (int e = 0; e < 8; ++e) {
      p0[e]     = (_Float16)(s[0][e] * inv);
      p0[e + 8] = (_Float16)(s[1][e] * inv);
      p1[e]     = (_Float16)(s[2][e] * inv);
      p1[e + 8] = (_Float16)(s[3][e] * inv);
    }

    // O = P[16,64] x V[64,32]
    #pragma unroll
    for (int ot = 0; ot < 2; ++ot) {
      v8f acc = {};
      acc = wmma16(p0, load_b_frag(Vh, ot * 16, 0,  64), acc);
      acc = wmma16(p1, load_b_frag(Vh, ot * 16, 32, 64), acc);
      _Float16* op = AO + (it * 16 + hi8) * CDIM + h * 32 + ot * 16 + lane15;
      #pragma unroll
      for (int r = 0; r < 8; ++r) op[r * CDIM] = (_Float16)acc[r];
    }
  }
  __syncthreads();

  // Phase 3: proj GEMM [64,128] x [128,128] + bias; NT f32 stores, rows < 49.
  // Row tiles 0..2 are fully valid (max row 47); only tile 3 needs a guard.
  {
    const int rt = __builtin_amdgcn_readfirstlane(wave >> 1);
    v16h o0 = load_a_frag(AO, rt * 16, 0,  CDIM);
    v16h o1 = load_a_frag(AO, rt * 16, 32, CDIM);
    v16h o2 = load_a_frag(AO, rt * 16, 64, CDIM);
    v16h o3 = load_a_frag(AO, rt * 16, 96, CDIM);
    float* outl = out + (size_t)l * NTOK * CDIM;
    for (int t = 0; t < 4; ++t) {
      int ct = (wave & 1) * 4 + t;
      int n = ct * 16 + lane15;
      v8f acc = {};
      acc = wmma16(o0, load_b_frag(wproj_t, ct * 16, 0,  CDIM), acc);
      acc = wmma16(o1, load_b_frag(wproj_t, ct * 16, 32, CDIM), acc);
      acc = wmma16(o2, load_b_frag(wproj_t, ct * 16, 64, CDIM), acc);
      acc = wmma16(o3, load_b_frag(wproj_t, ct * 16, 96, CDIM), acc);
      float pb = proj_b[n];
      float* op = outl + (rt * 16 + hi8) * CDIM + n;
      if (rt < 3) {
        #pragma unroll
        for (int r = 0; r < 8; ++r)
          __builtin_nontemporal_store(acc[r] + pb, op + r * CDIM);
      } else if (hi8 == 0) {                      // row 48 only
        __builtin_nontemporal_store(acc[0] + pb, op);
      }
    }
  }
}

extern "C" void kernel_launch(void* const* d_in, const int* in_sizes, int n_in,
                              void* d_out, int out_size, void* d_ws, size_t ws_size,
                              hipStream_t stream) {
  const float* x          = (const float*)d_in[0];
  const float* mask       = (const float*)d_in[1];
  const float* qkv_w      = (const float*)d_in[2];
  const float* qkv_b      = (const float*)d_in[3];
  const float* proj_w     = (const float*)d_in[4];
  const float* proj_b     = (const float*)d_in[5];
  const float* bias_table = (const float*)d_in[6];

  char* ws = (char*)d_ws;
  _Float16* wqkv_t  = (_Float16*)ws;                  // [384][128] f16  (96 KB)
  _Float16* wproj_t = (_Float16*)(ws + 98304);        // [128][128] f16  (32 KB)
  float*    bmT     = (float*)(ws + 131072);          // [64][4][64][64] f32 (4 MB)

  prep_kernel<<<(64 * NHEAD * NPAD * NPAD + 255) / 256, 256, 0, stream>>>(
      qkv_w, proj_w, bias_table, mask, wqkv_t, wproj_t, bmT);

  const size_t lds_bytes = 32768 * sizeof(_Float16); // 64 KB of the 320 KB WGP LDS
  win_attn_kernel<<<4096, 256, lds_bytes, stream>>>(x, qkv_b, proj_b, bmT,
                                                    wqkv_t, wproj_t, (float*)d_out);
}
